// SpatialAwareVAE_10574209482793
// MI455X (gfx1250) — matrix operand order repaired and checked by reading, not compiled, not run on hardware
//
#include <hip/hip_runtime.h>

#define BB 4
#define NN 3072
#define DD 128
#define HH 4
#define DK 32

typedef __attribute__((ext_vector_type(16))) __bf16 v16bf;
typedef __attribute__((ext_vector_type(16))) unsigned short v16u;
typedef __attribute__((ext_vector_type(8))) float v8f;

// native f32 -> bf16 (RNE) convert; pairs fuse into v_cvt_pk_bf16_f32
__device__ __forceinline__ unsigned short f2bf(float f) {
  return __builtin_bit_cast(unsigned short, static_cast<__bf16>(f));
}

__device__ __forceinline__ v8f v8f_zero() {
  v8f z;
#pragma unroll
  for (int i = 0; i < 8; ++i) z[i] = 0.f;
  return z;
}

__device__ __forceinline__ v8f wmma_bf16(v16u a, v16u b, v8f c) {
  return __builtin_amdgcn_wmma_f32_16x16x32_bf16(
      false, __builtin_bit_cast(v16bf, a),
      false, __builtin_bit_cast(v16bf, b),
      (short)0, c, false, false);
}

// ---------------------------------------------------------------------------
// h0 = x + sinusoidal PE (computed on the fly)
// ---------------------------------------------------------------------------
__global__ void add_pe_kernel(const float* __restrict__ x, float* __restrict__ y) {
  int idx = blockIdx.x * 256 + threadIdx.x;          // grid sized exactly
  int d = idx & (DD - 1);
  int n = (idx >> 7) % NN;                           // DD == 128
  float i2 = (float)(d & ~1);
  float freq = __expf(i2 * -0.07195578f);            // -ln(10000)/128
  float ang = (float)n * freq;
  float pe = (d & 1) ? __cosf(ang) : __sinf(ang);
  y[idx] = x[idx] + pe;
}

// ---------------------------------------------------------------------------
// Y[M x 128] = X[M x 128] @ W[128 x 128] (+ bias).
// Block = 64 rows x 64 cols (4 waves, wave -> 16-col tile).  W fragments for
// the wave's column tile are hoisted once (4 K-chunks) and reused across the
// 4 row tiles; all strided accesses use base-pointer + immediate offsets.
// grid (M/64, 2), block 128.
// ---------------------------------------------------------------------------
__global__ void __launch_bounds__(128)
gemm128_kernel(const float* __restrict__ X, const float* __restrict__ W,
               const float* __restrict__ bias, float* __restrict__ Y,
               int hasBias) {
  const int lane = threadIdx.x & 31;
  const int wave = threadIdx.x >> 5;
  const int l15 = lane & 15;
  const int half = lane >> 4;
  const int rt0 = blockIdx.x * 64;
  const int ct = (blockIdx.y * 4 + wave) * 16;

  // hoist the wave's B fragments: W[k][ct+l15] column, 4 chunks of K=32
  v16u b[4];
#pragma unroll
  for (int c = 0; c < 4; ++c) {
    const float* wp = W + (c * 32 + half * 16) * DD + ct + l15;
#pragma unroll
    for (int e = 0; e < 16; ++e) b[c][e] = f2bf(wp[e * DD]);
  }

  const float bcol = hasBias ? bias[ct + l15] : 0.f;

#pragma unroll
  for (int r = 0; r < 4; ++r) {
    const int rt = rt0 + r * 16;
    v8f acc = v8f_zero();
#pragma unroll
    for (int c = 0; c < 4; ++c) {
      const float* xp = X + (rt + l15) * DD + c * 32 + half * 16;
      v16u a;
#pragma unroll
      for (int e = 0; e < 8; ++e) {
        a[e]     = f2bf(xp[e]);
        a[e + 8] = f2bf(xp[e + 8]);
      }
      acc = wmma_bf16(a, b[c], acc);
    }
    float* yp = Y + (rt + half * 8) * DD + ct + l15;
#pragma unroll
    for (int v = 0; v < 8; ++v) yp[v * DD] = acc[v] + bcol;
  }
}

// ---------------------------------------------------------------------------
// Fused flash-attention for one (batch, head, 64-row q block).
// block = 128 threads (4 waves, each owns a 16-row q tile); K/V chunks of 32
// keys staged in LDS as bf16 (float4 load -> ushort4 b64 store); online
// softmax in registers; P tile routed C-layout -> A-layout through per-wave
// LDS with s_wait_dscnt.
// ---------------------------------------------------------------------------
__global__ void __launch_bounds__(128)
attn_kernel(const float* __restrict__ Q, const float* __restrict__ K,
            const float* __restrict__ V, float* __restrict__ O) {
  __shared__ __align__(16) unsigned short Ks[32][40];
  __shared__ __align__(16) unsigned short Vs[32][40];
  __shared__ __align__(16) unsigned short Ps[4][16][40];

  const int tid = threadIdx.x;
  const int lane = tid & 31;
  const int wave = tid >> 5;
  const int l15 = lane & 15;
  const int half = lane >> 4;

  int bid = blockIdx.x;
  const int qb = bid % (NN / 64);
  bid /= (NN / 64);
  const int hh = bid % HH;
  const int bb = bid / HH;

  const int qrow = qb * 64 + wave * 16 + l15;
  const float* qp = Q + (bb * NN + qrow) * DD + hh * DK + half * 16;

  v16u qa;
#pragma unroll
  for (int e = 0; e < 8; ++e) {
    qa[e]     = f2bf(qp[e] * 0.176776695f);            // 1/sqrt(dk)
    qa[e + 8] = f2bf(qp[e + 8] * 0.176776695f);
  }

  float m[8], l[8];
  v8f o0 = v8f_zero(), o1 = v8f_zero();
#pragma unroll
  for (int v = 0; v < 8; ++v) { m[v] = -3.0e38f; l[v] = 0.f; }

  for (int kc = 0; kc < NN; kc += 32) {
    __syncthreads();
#pragma unroll
    for (int t = tid; t < 256; t += 128) {               // 32 keys x 32 dims / 4
      int key = t >> 3, dc = (t & 7) * 4;
      int goff = (bb * NN + kc + key) * DD + hh * DK + dc;
      float4 k4 = *(const float4*)(K + goff);
      float4 v4 = *(const float4*)(V + goff);
      ushort4 ksv, vsv;
      ksv.x = f2bf(k4.x); ksv.y = f2bf(k4.y); ksv.z = f2bf(k4.z); ksv.w = f2bf(k4.w);
      vsv.x = f2bf(v4.x); vsv.y = f2bf(v4.y); vsv.z = f2bf(v4.z); vsv.w = f2bf(v4.w);
      *(ushort4*)&Ks[key][dc] = ksv;
      *(ushort4*)&Vs[key][dc] = vsv;
    }
    __syncthreads();

    v16u kb0, kb1;
#pragma unroll
    for (int e = 0; e < 16; ++e) {
      int kd = half * 16 + e;
      kb0[e] = Ks[l15][kd];
      kb1[e] = Ks[16 + l15][kd];
    }
    v8f s0 = wmma_bf16(qa, kb0, v8f_zero());
    v8f s1 = wmma_bf16(qa, kb1, v8f_zero());

#pragma unroll
    for (int v = 0; v < 8; ++v) {
      float mx = fmaxf(s0[v], s1[v]);
      mx = fmaxf(mx, __shfl_xor(mx, 1, 32));
      mx = fmaxf(mx, __shfl_xor(mx, 2, 32));
      mx = fmaxf(mx, __shfl_xor(mx, 4, 32));
      mx = fmaxf(mx, __shfl_xor(mx, 8, 32));
      float nm = fmaxf(m[v], mx);
      float p0 = __expf(s0[v] - nm);
      float p1 = __expf(s1[v] - nm);
      float rs = p0 + p1;
      rs += __shfl_xor(rs, 1, 32);
      rs += __shfl_xor(rs, 2, 32);
      rs += __shfl_xor(rs, 4, 32);
      rs += __shfl_xor(rs, 8, 32);
      float sc = __expf(m[v] - nm);
      l[v] = l[v] * sc + rs;
      m[v] = nm;
      o0[v] *= sc;
      o1[v] *= sc;
      int pr = half * 8 + v;
      Ps[wave][pr][l15] = f2bf(p0);
      Ps[wave][pr][16 + l15] = f2bf(p1);
    }
    asm volatile("s_wait_dscnt 0" ::: "memory");   // wave-local LDS RAW

    v16u pa, vb0, vb1;
#pragma unroll
    for (int e = 0; e < 8; ++e) {
      pa[e]     = Ps[wave][l15][half * 16 + e];
      pa[e + 8] = Ps[wave][l15][half * 16 + 8 + e];
    }
#pragma unroll
    for (int e = 0; e < 16; ++e) {
      int key = half * 16 + e;
      vb0[e] = Vs[key][l15];
      vb1[e] = Vs[key][16 + l15];
    }
    o0 = wmma_bf16(pa, vb0, o0);
    o1 = wmma_bf16(pa, vb1, o1);
  }

  float* op = O + (bb * NN + qb * 64 + wave * 16 + half * 8) * DD + hh * DK + l15;
#pragma unroll
  for (int v = 0; v < 8; ++v) {
    float inv = 1.0f / l[v];
    op[v * DD]      = o0[v] * inv;
    op[v * DD + 16] = o1[v] * inv;
  }
}

// ---------------------------------------------------------------------------
// Fused graph conv aggregation: Y = norm_adj @ Hm, adj = exp(-d2/(2*0.1^2)).
// Gaussian weights computed elementwise directly in WMMA A-fragment layout.
// Positions chunk staged with CDNA5 async global->LDS DMA (ASYNCcnt path);
// Hm chunk staged as bf16 via float4/ushort4.  block = 128, grid = B*(N/64).
// ---------------------------------------------------------------------------
__global__ void __launch_bounds__(128)
gconv_kernel(const float* __restrict__ pos, const float* __restrict__ Hm,
             float* __restrict__ Y) {
  __shared__ __align__(16) unsigned short Hs[32][136];
  __shared__ __align__(16) float Pp[32][3];
  __shared__ float Rs[4][16];

  const int tid = threadIdx.x;
  const int lane = tid & 31;
  const int wave = tid >> 5;
  const int l15 = lane & 15;
  const int half = lane >> 4;

  int bid = blockIdx.x;
  const int rb = bid % (NN / 64);
  const int bb = bid / (NN / 64);
  const int row = rb * 64 + wave * 16 + l15;

  const float px = pos[(bb * NN + row) * 3 + 0];
  const float py = pos[(bb * NN + row) * 3 + 1];
  const float pz = pos[(bb * NN + row) * 3 + 2];

  v8f acc[8];
#pragma unroll
  for (int t = 0; t < 8; ++t) acc[t] = v8f_zero();
  float rsum = 0.f;

  for (int kc = 0; kc < NN; kc += 32) {
    __syncthreads();
    // async DMA the fp32 positions chunk straight into LDS (no VGPR round trip)
    if (tid < 96) {
      unsigned lp = (unsigned)(size_t)(&Pp[0][0]) + (unsigned)tid * 4u;
      const float* gp = pos + (bb * NN + kc) * 3 + tid;
      asm volatile("global_load_async_to_lds_b32 %0, %1, off"
                   :: "v"(lp), "v"(gp) : "memory");
    }
#pragma unroll
    for (int t = tid; t < 1024; t += 128) {              // 32 rows x 128 cols / 4
      int r = t >> 5, c = (t & 31) * 4;
      float4 h4 = *(const float4*)(Hm + (bb * NN + kc + r) * DD + c);
      ushort4 hsv;
      hsv.x = f2bf(h4.x); hsv.y = f2bf(h4.y); hsv.z = f2bf(h4.z); hsv.w = f2bf(h4.w);
      *(ushort4*)&Hs[r][c] = hsv;
    }
    asm volatile("s_wait_asynccnt 0" ::: "memory");
    __syncthreads();

    v16u wa;
#pragma unroll
    for (int e = 0; e < 16; ++e) {
      int mi = half * 16 + e;
      float dx = px - Pp[mi][0];
      float dy = py - Pp[mi][1];
      float dz = pz - Pp[mi][2];
      float w = __expf((dx * dx + dy * dy + dz * dz) * -50.0f);  // -1/(2*sigma^2)
      rsum += w;
      wa[e] = f2bf(w);
    }
#pragma unroll
    for (int t = 0; t < 8; ++t) {
      v16u hb;
#pragma unroll
      for (int e = 0; e < 16; ++e) hb[e] = Hs[half * 16 + e][t * 16 + l15];
      acc[t] = wmma_bf16(wa, hb, acc[t]);
    }
  }

  rsum += __shfl_xor(rsum, 16, 32);          // combine the two half-wave lanes
  if (lane < 16) Rs[wave][l15] = rsum;
  asm volatile("s_wait_dscnt 0" ::: "memory");

  float* yp = Y + (bb * NN + rb * 64 + wave * 16 + half * 8) * DD + l15;
#pragma unroll
  for (int v = 0; v < 8; ++v) {
    float inv = 1.0f / (Rs[wave][half * 8 + v] + 1e-8f);
#pragma unroll
    for (int t = 0; t < 8; ++t)
      yp[v * DD + t * 16] = acc[t][v] * inv;
  }
}

// ---------------------------------------------------------------------------
extern "C" void kernel_launch(void* const* d_in, const int* in_sizes, int n_in,
                              void* d_out, int out_size, void* d_ws, size_t ws_size,
                              hipStream_t stream) {
  const float* x     = (const float*)d_in[0];
  const float* pos   = (const float*)d_in[1];
  const float* sa_Wq = (const float*)d_in[2];
  const float* sa_bq = (const float*)d_in[3];
  const float* sa_Wk = (const float*)d_in[4];
  const float* sa_bk = (const float*)d_in[5];
  const float* sa_Wv = (const float*)d_in[6];
  const float* sa_bv = (const float*)d_in[7];
  const float* sa_Wo = (const float*)d_in[8];
  const float* sa_bo = (const float*)d_in[9];
  const float* gc_W  = (const float*)d_in[10];
  const float* da_Wq = (const float*)d_in[11];
  const float* da_bq = (const float*)d_in[12];
  const float* da_Wk = (const float*)d_in[13];
  const float* da_bk = (const float*)d_in[14];
  const float* da_Wv = (const float*)d_in[15];
  const float* da_bv = (const float*)d_in[16];
  const float* da_Wo = (const float*)d_in[17];
  const float* da_bo = (const float*)d_in[18];

  const size_t SLOT = (size_t)BB * NN * DD;       // 1.57M floats / slot
  float* s0 = (float*)d_ws;
  float* s1 = s0 + SLOT;
  float* s2 = s1 + SLOT;
  float* s3 = s2 + SLOT;
  float* s4 = s3 + SLOT;

  const dim3 gg(BB * NN / 64, 2);                 // GEMM grid (64x64 per block)
  const dim3 ga(BB * HH * (NN / 64));             // attention grid
  const dim3 gc(BB * (NN / 64));                  // graph-conv grid

  add_pe_kernel<<<dim3(BB * NN * DD / 256), 256, 0, stream>>>(x, s0);

  // MHA 1
  gemm128_kernel<<<gg, 128, 0, stream>>>(s0, sa_Wq, sa_bq, s1, 1);
  gemm128_kernel<<<gg, 128, 0, stream>>>(s0, sa_Wk, sa_bk, s2, 1);
  gemm128_kernel<<<gg, 128, 0, stream>>>(s0, sa_Wv, sa_bv, s3, 1);
  attn_kernel<<<ga, 128, 0, stream>>>(s1, s2, s3, s4);
  gemm128_kernel<<<gg, 128, 0, stream>>>(s4, sa_Wo, sa_bo, s0, 1);

  // graph conv
  gemm128_kernel<<<gg, 128, 0, stream>>>(s0, gc_W, nullptr, s1, 0);
  gconv_kernel<<<gc, 128, 0, stream>>>(pos, s1, s2);

  // MHA 2
  gemm128_kernel<<<gg, 128, 0, stream>>>(s2, da_Wq, da_bq, s3, 1);
  gemm128_kernel<<<gg, 128, 0, stream>>>(s2, da_Wk, da_bk, s4, 1);
  gemm128_kernel<<<gg, 128, 0, stream>>>(s2, da_Wv, da_bv, s0, 1);
  attn_kernel<<<ga, 128, 0, stream>>>(s3, s4, s0, s1);
  gemm128_kernel<<<gg, 128, 0, stream>>>(s1, da_Wo, da_bo, (float*)d_out, 1);
}
